// encoder_86131274154331
// MI455X (gfx1250) — compile-verified, matrix operand order checked by reading, not run on hardware
//
#include <hip/hip_runtime.h>
#include <hip/hip_bf16.h>

#define SEQ 200
#define EMB 1024
#define NHEAD 8
#define DHEAD 128
#define DFF 2048
#define NLAYER 5

typedef __attribute__((ext_vector_type(16))) __bf16        bf16x16;
typedef __attribute__((ext_vector_type(8)))  float         f32x8;
typedef __attribute__((ext_vector_type(8)))  unsigned int  u32x8;

// ---------- Tensor Data Mover (gfx1250 TDM), compile-safe dispatch ----------
#if __has_builtin(__builtin_amdgcn_tensor_load_to_lds)
#define HAVE_TDM 1
typedef __attribute__((ext_vector_type(4))) unsigned int u32x4;
typedef __attribute__((ext_vector_type(8))) int          i32x8;
typedef __attribute__((ext_vector_type(4))) int          i32x4;

// 2-D fp32 tile load: global (row_stride elements/row) -> LDS at lds_off,
// LDS rows padded via D# pad fields (pad every 2^(pad_code+1) dwords by pad_dw).
// tensor_h < tile_h zero-fills OOB rows in hardware (M-edge handling).
__device__ __forceinline__ void tdm_load_2d(
    const float* gsrc, unsigned lds_off,
    unsigned tensor_w, unsigned tensor_h,
    unsigned tile_w, unsigned tile_h,
    unsigned row_stride, unsigned pad_code, unsigned pad_dw)
{
  unsigned long long ga = (unsigned long long)gsrc;
  u32x4 g0;
  g0.x = 1u;                                        // count=1 (valid user D#)
  g0.y = lds_off;                                   // lds_addr
  g0.z = (unsigned)ga;                              // global_addr[31:0]
  g0.w = (unsigned)((ga >> 32) & 0x01FFFFFFu) | (2u << 30);  // addr[56:32] | type=2
  i32x8 g1 = {
    (int)((2u << 16) | (1u << 20) | (pad_code << 22) | ((pad_dw - 1u) << 25)),
    (int)((tensor_w & 0xFFFFu) << 16),                               // dim0[15:0]
    (int)(((tensor_w >> 16) & 0xFFFFu) | ((tensor_h & 0xFFFFu) << 16)),
    (int)(((tensor_h >> 16) & 0xFFFFu) | ((tile_w & 0xFFFFu) << 16)),
    (int)(tile_h & 0xFFFFu),                                         // tile_dim1 (dim2=0)
    (int)row_stride,                                                 // dim0_stride[31:0]
    0, 0
  };
  i32x4 z4 = {0, 0, 0, 0};
#if defined(__clang_major__) && __clang_major__ >= 23
  i32x8 z8 = {0, 0, 0, 0, 0, 0, 0, 0};
  __builtin_amdgcn_tensor_load_to_lds(g0, g1, z4, z4, z8, 0);
#else
  __builtin_amdgcn_tensor_load_to_lds(g0, g1, z4, z4, 0);
#endif
}
#else
#define HAVE_TDM 0
#endif

// ---------- f32 -> packed bf16 (RNE) ----------
#if __has_builtin(__builtin_amdgcn_cvt_pk_bf16_f32)
typedef __attribute__((ext_vector_type(2))) __bf16 bf16x2;
__device__ __forceinline__ unsigned pack_bf(float lo, float hi) {
  bf16x2 p = __builtin_amdgcn_cvt_pk_bf16_f32(lo, hi);   // 1x v_cvt_pk_bf16_f32
  return __builtin_bit_cast(unsigned, p);
}
#else
// CDNA5 VOP3P opcodes 62/63: fused f32 FMA + RNE-convert into bf16 half-dword.
// bf16(x) = fma_mix(x, 1.0, 0): packs one dword in 2 VALU ops (vs 4 for bit math).
__device__ __forceinline__ unsigned pack_bf(float lo, float hi) {
  unsigned d;
  asm("v_fma_mixlo_bf16 %0, %1, 1.0, 0\n\t"
      "v_fma_mixhi_bf16 %0, %2, 1.0, 0"
      : "=&v"(d)
      : "v"(lo), "v"(hi));
  return d;
}
#endif

// ---------- block reductions ----------
__device__ __forceinline__ float block_sum(float v, float* red) {
  int tid = threadIdx.x;
  red[tid] = v;  __syncthreads();
  for (int off = 128; off > 0; off >>= 1) {
    if (tid < off) red[tid] += red[tid + off];
    __syncthreads();
  }
  float r = red[0];  __syncthreads();
  return r;
}
__device__ __forceinline__ float block_max(float v, float* red) {
  int tid = threadIdx.x;
  red[tid] = v;  __syncthreads();
  for (int off = 128; off > 0; off >>= 1) {
    if (tid < off) red[tid] = fmaxf(red[tid], red[tid + off]);
    __syncthreads();
  }
  float r = red[0];  __syncthreads();
  return r;
}

// ---------- fragment builders: fp32 LDS -> packed bf16 WMMA fragments ----------
// CDNA5 16-bit A/B layout: lane%16 = M/N row, lane/16 = K-half,
// VGPR v holds packed K pair at koff = (v/4)*16 + half*8 + (v%4)*2.
#define AST 36   // A LDS row stride (floats): 144B, 16B-aligned, conflict-free
#define BST 68   // B LDS row stride (floats): 272B, 16B-aligned

__device__ __forceinline__ bf16x16 frag_a(const float* row, int half) {
  u32x8 r;
#pragma unroll
  for (int v = 0; v < 8; ++v) {
    int koff = ((v >> 2) << 4) | (half << 3) | ((v & 3) << 1);
    float2 f = *(const float2*)(row + koff);
    r[v] = pack_bf(f.x, f.y);
  }
  return __builtin_bit_cast(bf16x16, r);
}
__device__ __forceinline__ bf16x16 frag_b(const float* base, int n, int half) {
  u32x8 r;
#pragma unroll
  for (int v = 0; v < 8; ++v) {
    int k = ((v >> 2) << 4) | (half << 3) | ((v & 3) << 1);
    r[v] = pack_bf(base[k * BST + n], base[(k + 1) * BST + n]);
  }
  return __builtin_bit_cast(bf16x16, r);
}

// ---------- GEMM: C[M,N] = relu?(A[M,K] @ B[K,N] + bias[N]) ----------
// 128 threads = 4 waves; block tile 32(M) x 64(N), BK=32; each wave: 16x32 (2 wmma/step).
// Double-buffered LDS: the Tensor Data Mover streams tile t+1 while tile t is in the
// WMMAs; s_wait_tensorcnt(2) retires the older DMA pair (TDM is in-order per wave).
__global__ __launch_bounds__(128) void gemm_kernel(
    const float* __restrict__ A, const float* __restrict__ B,
    const float* __restrict__ bias, float* __restrict__ C,
    int M, int N, int K, int relu)
{
  __shared__ __align__(16) float sA[2][32 * AST];
  __shared__ __align__(16) float sB[2][32 * BST];

  const int tid    = threadIdx.x;
  const int wave   = tid >> 5;
  const int lane   = tid & 31;
  const int lane16 = lane & 15;
  const int half   = lane >> 4;
  const int tm     = (wave & 1) << 4;    // 0 or 16
  const int tn     = (wave >> 1) << 5;   // 0 or 32
  const int bm     = blockIdx.x * 32;
  const int bn     = blockIdx.y * 64;

  f32x8 acc0, acc1;
#pragma unroll
  for (int v = 0; v < 8; ++v) { acc0[v] = 0.0f; acc1[v] = 0.0f; }

#if HAVE_TDM
  const unsigned sAo[2] = { (unsigned)(unsigned long long)(void*)&sA[0][0],
                            (unsigned)(unsigned long long)(void*)&sA[1][0] };
  const unsigned sBo[2] = { (unsigned)(unsigned long long)(void*)&sB[0][0],
                            (unsigned)(unsigned long long)(void*)&sB[1][0] };
#endif

  // tile loader: Tensor DMA when available, vectorized loads otherwise
  auto load_tiles = [&](int buf, int k0) {
#if HAVE_TDM
    if (tid < 32) {   // wave 0 drives the TDM (EXEC-independent DMA)
      tdm_load_2d(A + (size_t)bm * K + k0, sAo[buf],
                  (unsigned)(K - k0), (unsigned)(M - bm), 32u, 32u,
                  (unsigned)K, 4u /*pad every 32dw*/, 4u /*by 4dw -> stride 36*/);
      tdm_load_2d(B + (size_t)k0 * N + bn, sBo[buf],
                  (unsigned)(N - bn), (unsigned)(K - k0), 64u, 32u,
                  (unsigned)N, 5u /*pad every 64dw*/, 4u /*-> stride 68*/);
    }
#else
    // A tile 32x32: branch-free (clamped row), 128-bit traffic
#pragma unroll
    for (int it = 0; it < 2; ++it) {
      int idx = tid + (it << 7);
      int r = idx >> 3, c4 = idx & 7;
      int ar = bm + r; ar = ar < M ? ar : M - 1;
      *(float4*)(sA[buf] + r * AST + (c4 << 2)) =
          *(const float4*)(A + (size_t)ar * K + k0 + (c4 << 2));
    }
    // B tile 32x64
#pragma unroll
    for (int it = 0; it < 4; ++it) {
      int idx = tid + (it << 7);
      int r = idx >> 4, c4 = idx & 15;
      *(float4*)(sB[buf] + r * BST + (c4 << 2)) =
          *(const float4*)(B + (size_t)(k0 + r) * N + bn + (c4 << 2));
    }
    if (k0 + 32 < K)   // global_prefetch_b8 of the following B tile into GL2
      __builtin_prefetch(B + (size_t)(k0 + 32 + (tid >> 4)) * N + bn + ((tid & 15) << 2), 0, 1);
#endif
  };

  load_tiles(0, 0);
  int buf = 0;
  for (int k0 = 0; k0 < K; k0 += 32, buf ^= 1) {
    const bool more = (k0 + 32) < K;
    if (more) load_tiles(buf ^ 1, k0 + 32);   // overlap DMA of next tile with math
#if HAVE_TDM
    if (tid < 32) {
      if (more) __builtin_amdgcn_s_wait_tensorcnt(2);  // older pair (current tile) done
      else      __builtin_amdgcn_s_wait_tensorcnt(0);
    }
#endif
    __syncthreads();

    bf16x16 af = frag_a(sA[buf] + (tm + lane16) * AST, half);
    bf16x16 b0 = frag_b(sB[buf], tn + lane16, half);
    bf16x16 b1 = frag_b(sB[buf], tn + 16 + lane16, half);
    acc0 = __builtin_amdgcn_wmma_f32_16x16x32_bf16(false, af, false, b0,
                                                   (short)0, acc0, false, false);
    acc1 = __builtin_amdgcn_wmma_f32_16x16x32_bf16(false, af, false, b1,
                                                   (short)0, acc1, false, false);
    __syncthreads();
  }

  // C layout: lane%16 = N col, VGPR v -> row M = v + 8*(lane/16)
  const int gn0 = bn + tn + lane16;
  const float bb0 = bias[gn0];
  const float bb1 = bias[gn0 + 16];
#pragma unroll
  for (int v = 0; v < 8; ++v) {
    int gm = bm + tm + v + (half << 3);
    if (gm < M) {
      float o0 = acc0[v] + bb0;
      float o1 = acc1[v] + bb1;
      if (relu) { o0 = fmaxf(o0, 0.0f); o1 = fmaxf(o1, 0.0f); }
      C[(size_t)gm * N + gn0]      = o0;
      C[(size_t)gm * N + gn0 + 16] = o1;
    }
  }
}

// ---------- embedding + sinusoidal positional encoding ----------
__global__ __launch_bounds__(256) void embed_pe_kernel(
    const int* __restrict__ x, const float* __restrict__ embed,
    float* __restrict__ h)
{
  int s = blockIdx.x;
  int token = x[s];
  const float nlog1e4 = -logf(10000.0f);
  for (int e = threadIdx.x; e < EMB; e += 256) {
    int j2 = e & ~1;
    float freq = expf(nlog1e4 * (float)j2 / (float)EMB);
    float arg  = (float)s * freq;
    float pe   = (e & 1) ? cosf(arg) : sinf(arg);
    h[(size_t)s * EMB + e] = embed[(size_t)token * EMB + e] + pe;
  }
}

// ---------- LayerNorm (optionally fused residual add), one row/block ----------
__global__ __launch_bounds__(256) void ln_kernel(
    const float* __restrict__ x, const float* __restrict__ res,
    float* __restrict__ y, const float* __restrict__ g,
    const float* __restrict__ b)
{
  __shared__ float red[256];
  int s = blockIdx.x, tid = threadIdx.x;
  float vals[EMB / 256];
  float sum = 0.0f;
#pragma unroll
  for (int i = 0; i < EMB / 256; ++i) {
    int e = tid + i * 256;
    float v = x[(size_t)s * EMB + e];
    if (res) v += res[(size_t)s * EMB + e];
    vals[i] = v;  sum += v;
  }
  float mean = block_sum(sum, red) / (float)EMB;
  float s2 = 0.0f;
#pragma unroll
  for (int i = 0; i < EMB / 256; ++i) { float d = vals[i] - mean; s2 += d * d; }
  float var = block_sum(s2, red) / (float)EMB;
  float inv = rsqrtf(var + 1e-5f);
#pragma unroll
  for (int i = 0; i < EMB / 256; ++i) {
    int e = tid + i * 256;
    y[(size_t)s * EMB + e] = (vals[i] - mean) * inv * g[e] + b[e];
  }
}

// ---------- GroupAttention: only the two score diagonals are ever used ----------
__global__ __launch_bounds__(256) void diag_scores_kernel(
    const float* __restrict__ q, const float* __restrict__ k,
    float* __restrict__ sup, float* __restrict__ sub)
{
  __shared__ float red[256];
  int s = blockIdx.x, tid = threadIdx.x;
  float a1 = 0.0f, a2 = 0.0f;
  for (int e = tid; e < EMB; e += 256) {
    float qe = q[(size_t)s * EMB + e];
    if (s + 1 < SEQ) a1 += qe * k[(size_t)(s + 1) * EMB + e];
    if (s >= 1)      a2 += qe * k[(size_t)(s - 1) * EMB + e];
  }
  float r1 = block_sum(a1, red);
  float r2 = block_sum(a2, red);
  if (tid == 0) { sup[s] = r1 / (float)EMB; sub[s] = r2 / (float)EMB; }
}

// tri/log/exp chain collapses to prefix sums: g[i,j] = exp(C[j]-C[i]) for j>=i.
__global__ __launch_bounds__(256) void group_build_kernel(
    const float* __restrict__ sup, const float* __restrict__ sub,
    const int* __restrict__ mask, float* __restrict__ gp)
{
  __shared__ float nsup[SEQ], nsub[SEQ], Lar[SEQ], Cs[SEQ], symsub[SEQ];
  int tid = threadIdx.x;
  for (int q = tid; q < SEQ; q += 256) {
    bool vs = (q + 1 < SEQ) && (mask[q + 1] != 0);
    bool vb = (q >= 1)      && (mask[q - 1] != 0);
    float ss = vs ? sup[q] : -1e9f;
    float sb = vb ? sub[q] : -1e9f;
    float m  = fmaxf(ss, sb);
    float es = expf(ss - m), eb = expf(sb - m);
    float eo = expf(-1e9f - m) * (float)(SEQ - 2);  // other row entries (all -1e9)
    float den = es + eb + eo;
    nsup[q] = es / den;
    nsub[q] = eb / den;
  }
  __syncthreads();
  for (int q = tid; q < SEQ; q += 256) {
    if (q + 1 < SEQ) {
      float v = sqrtf(nsup[q] * nsub[q + 1] + 1e-9f);  // symmetrized superdiag
      Lar[q] = logf(v + 1e-9f);
    } else Lar[q] = 0.0f;
    symsub[q] = (q >= 1) ? sqrtf(nsub[q] * nsup[q - 1] + 1e-9f) : 0.0f;
  }
  __syncthreads();
  if (tid == 0) {
    Cs[0] = 0.0f;
    for (int i = 1; i < SEQ; ++i) Cs[i] = Cs[i - 1] + Lar[i - 1];
  }
  __syncthreads();
  for (int idx = tid; idx < SEQ * SEQ; idx += 256) {
    int i = idx / SEQ, j = idx % SEQ;
    float gij = (j >= i - 1) ? ((j > i) ? expf(Cs[j] - Cs[i]) : 1.0f) : 0.0f;
    float gji = (i >= j - 1) ? ((i > j) ? expf(Cs[i] - Cs[j]) : 1.0f) : 0.0f;
    float ex  = (j == i - 1) ? symsub[i] : 1e-9f;
    gp[idx] = gij + gji + ex;
  }
}

// ---------- fused MHA row kernel: scores -> softmax*group_prob -> @V ----------
__global__ __launch_bounds__(256) void attn_kernel(
    const float* __restrict__ Q, const float* __restrict__ K,
    const float* __restrict__ V, const float* __restrict__ gp,
    const int* __restrict__ mask, float* __restrict__ Z)
{
  __shared__ float qsh[DHEAD];
  __shared__ float sc[SEQ];
  __shared__ float red[256];
  int qi = blockIdx.x, hd = blockIdx.y, tid = threadIdx.x;
  int hb = hd * DHEAD;
  if (tid < DHEAD) qsh[tid] = Q[(size_t)qi * EMB + hb + tid];
  __syncthreads();

  const float scale = 0.0883883476483184f;  // 1/sqrt(128)
  for (int k = tid; k < SEQ; k += 256) {
    const float* kr = K + (size_t)k * EMB + hb;
    float a = 0.0f;
#pragma unroll 8
    for (int d = 0; d < DHEAD; ++d) a += qsh[d] * kr[d];
    sc[k] = mask[k] ? a * scale : -1e9f;
  }
  __syncthreads();

  float m = -3.4e38f;
  for (int k = tid; k < SEQ; k += 256) m = fmaxf(m, sc[k]);
  m = block_max(m, red);
  float sum = 0.0f;
  for (int k = tid; k < SEQ; k += 256) { float e = expf(sc[k] - m); sc[k] = e; sum += e; }
  sum = block_sum(sum, red);
  float inv = 1.0f / sum;
  for (int k = tid; k < SEQ; k += 256) sc[k] = sc[k] * inv * gp[(size_t)qi * SEQ + k];
  __syncthreads();

  for (int d = tid; d < DHEAD; d += 256) {
    float acc = 0.0f;
    for (int k = 0; k < SEQ; ++k) acc += sc[k] * V[(size_t)k * EMB + hb + d];
    Z[(size_t)qi * EMB + hb + d] = acc;
  }
}

// ---------- driver ----------
extern "C" void kernel_launch(void* const* d_in, const int* in_sizes, int n_in,
                              void* d_out, int out_size, void* d_ws, size_t ws_size,
                              hipStream_t stream) {
  (void)in_sizes; (void)n_in; (void)out_size; (void)ws_size;
  const int*   x     = (const int*)d_in[0];
  const int*   mask  = (const int*)d_in[1];
  const float* embed = (const float*)d_in[2];
  const float* gq_w  = (const float*)d_in[3];
  const float* gq_b  = (const float*)d_in[4];
  const float* gk_w  = (const float*)d_in[5];
  const float* gk_b  = (const float*)d_in[6];
  const float* gn_g  = (const float*)d_in[7];
  const float* gn_b  = (const float*)d_in[8];
  const float* wq    = (const float*)d_in[9];
  const float* wq_b  = (const float*)d_in[10];
  const float* wk    = (const float*)d_in[11];
  const float* wk_b  = (const float*)d_in[12];
  const float* wv    = (const float*)d_in[13];
  const float* wv_b  = (const float*)d_in[14];
  const float* wo    = (const float*)d_in[15];
  const float* wo_b  = (const float*)d_in[16];
  const float* an_g  = (const float*)d_in[17];
  const float* an_b  = (const float*)d_in[18];
  const float* f1_w  = (const float*)d_in[19];
  const float* f1_b  = (const float*)d_in[20];
  const float* f2_w  = (const float*)d_in[21];
  const float* f2_b  = (const float*)d_in[22];
  const float* fn_g  = (const float*)d_in[23];
  const float* fn_b  = (const float*)d_in[24];

  float* ws = (float*)d_ws;
  const size_t SE = (size_t)SEQ * EMB;
  float* h   = ws;  ws += SE;
  float* qn  = ws;  ws += SE;
  float* qv  = ws;  ws += SE;   // group-attn query
  float* kv  = ws;  ws += SE;   // group-attn key
  float* Qb  = ws;  ws += SE;
  float* Kb  = ws;  ws += SE;
  float* Vb  = ws;  ws += SE;
  float* Zb  = ws;  ws += SE;
  float* tmp = ws;  ws += SE;
  float* ff  = ws;  ws += (size_t)SEQ * DFF;
  float* gp  = ws;  ws += (size_t)SEQ * SEQ;
  float* sup = ws;  ws += SEQ;
  float* sub = ws;  ws += SEQ;

  dim3 blk(256);
  dim3 gblk(128);
  dim3 gEE((SEQ + 31) / 32, EMB / 64);
  dim3 gEF((SEQ + 31) / 32, DFF / 64);

  embed_pe_kernel<<<SEQ, blk, 0, stream>>>(x, embed, h);

  for (int i = 0; i < NLAYER; ++i) {
    size_t oEE = (size_t)i * EMB * EMB;
    size_t oE  = (size_t)i * EMB;
    size_t oEF = (size_t)i * EMB * DFF;
    size_t oF  = (size_t)i * DFF;

    // GroupAttention (only diagonal scores needed)
    ln_kernel<<<SEQ, blk, 0, stream>>>(h, nullptr, qn, gn_g + oE, gn_b + oE);
    gemm_kernel<<<gEE, gblk, 0, stream>>>(qn, gq_w + oEE, gq_b + oE, qv, SEQ, EMB, EMB, 0);
    gemm_kernel<<<gEE, gblk, 0, stream>>>(qn, gk_w + oEE, gk_b + oE, kv, SEQ, EMB, EMB, 0);
    diag_scores_kernel<<<SEQ, blk, 0, stream>>>(qv, kv, sup, sub);
    group_build_kernel<<<1, blk, 0, stream>>>(sup, sub, mask, gp);

    // Multi-head attention
    gemm_kernel<<<gEE, gblk, 0, stream>>>(h, wq + oEE, wq_b + oE, Qb, SEQ, EMB, EMB, 0);
    gemm_kernel<<<gEE, gblk, 0, stream>>>(h, wk + oEE, wk_b + oE, Kb, SEQ, EMB, EMB, 0);
    gemm_kernel<<<gEE, gblk, 0, stream>>>(h, wv + oEE, wv_b + oE, Vb, SEQ, EMB, EMB, 0);
    attn_kernel<<<dim3(SEQ, NHEAD), blk, 0, stream>>>(Qb, Kb, Vb, gp, mask, Zb);
    gemm_kernel<<<gEE, gblk, 0, stream>>>(Zb, wo + oEE, wo_b + oE, tmp, SEQ, EMB, EMB, 0);
    ln_kernel<<<SEQ, blk, 0, stream>>>(h, tmp, h, an_g + oE, an_b + oE);

    // FFN
    gemm_kernel<<<gEF, gblk, 0, stream>>>(h, f1_w + oEF, f1_b + oF, ff, SEQ, DFF, EMB, 1);
    gemm_kernel<<<gEE, gblk, 0, stream>>>(ff, f2_w + oEF, f2_b + oE, tmp, SEQ, EMB, DFF, 0);
    ln_kernel<<<SEQ, blk, 0, stream>>>(h, tmp, h, fn_g + oE, fn_b + oE);
  }

  hipMemcpyAsync(d_out, h, SE * sizeof(float), hipMemcpyDeviceToDevice, stream);
}
